// GCNNet_63402307224304
// MI455X (gfx1250) — matrix operand order repaired and checked by reading, not compiled, not run on hardware
//
#include <hip/hip_runtime.h>
#include <hip/hip_bf16.h>

typedef __attribute__((ext_vector_type(16))) __bf16 v16bf;
typedef __attribute__((ext_vector_type(8)))  __bf16 v8bf;
typedef __attribute__((ext_vector_type(8)))  float  v8f;

#define WAVE 32
#define GEMM_WAVES 4
#define NT 4            // 16x64 output tile per wave

// ---------------------------------------------------------------- utilities
__global__ void zero_f32(float* __restrict__ p, size_t n) {
  size_t i = (size_t)blockIdx.x * blockDim.x + threadIdx.x;
  size_t stride = (size_t)gridDim.x * blockDim.x;
  for (; i < n; i += stride) p[i] = 0.0f;
}

__global__ void count_deg(const int* __restrict__ dst, float* __restrict__ deg, int E) {
  int i = blockIdx.x * blockDim.x + threadIdx.x;
  if (i < E) atomicAdd(&deg[dst[i]], 1.0f);
}

__global__ void make_dinv(float* __restrict__ deg, int N) {
  int i = blockIdx.x * blockDim.x + threadIdx.x;
  if (i < N) deg[i] = rsqrtf(deg[i] + 1.0f);   // +1 self-loop; always >= 1
}

// fp32 (row r, cols [0,cols), stride srcStride) -> bf16 (r, colsP, zero padded)
// 2-D launch: blockIdx.y = row, x-threads cover colsP. No div/mod.
__global__ void cvt_pad_bf16(const float* __restrict__ src, __bf16* __restrict__ dst,
                             int cols, int srcStride, int colsP) {
  int r = blockIdx.y;
  int c = blockIdx.x * blockDim.x + threadIdx.x;
  if (c >= colsP) return;
  float v = (c < cols) ? src[(size_t)r * srcStride + c] : 0.0f;
  dst[(size_t)r * colsP + c] = (__bf16)v;
}

// W (K x Nc fp32) -> Wt (Np x Kp bf16, zero padded) == W^T.  blockIdx.y = n.
__global__ void transpose_w_bf16(const float* __restrict__ W, __bf16* __restrict__ Wt,
                                 int K, int Nc, int Kp) {
  int n = blockIdx.y;
  int k = blockIdx.x * blockDim.x + threadIdx.x;
  if (k >= Kp) return;
  float v = (n < Nc && k < K) ? W[(size_t)k * Nc + n] : 0.0f;
  Wt[(size_t)n * Kp + k] = (__bf16)v;
}

// ---------------------------------------------- bf16 WMMA GEMM:  H = A @ W
// A:  M x KP bf16 row-major (KP % 32 == 0, zero padded)
// Bt: Np x KP bf16 row-major (= W^T, Np % 64 == 0, zero padded)
// H:  M x Np fp32 row-major
// Fragments are assembled directly from contiguous 16B global loads per the
// ISA 7.12.2 wave32 layouts -- no LDS, no divergent control flow.
template <int KP>
__global__ __launch_bounds__(GEMM_WAVES * WAVE)
void wmma_gemm_tn(const __bf16* __restrict__ A, const __bf16* __restrict__ Bt,
                  float* __restrict__ H, int M, int Np) {
  const int lane = threadIdx.x & 31;
  const int w    = threadIdx.x >> 5;
  const int row0 = blockIdx.x * 16;
  const int n0   = (blockIdx.y * GEMM_WAVES + w) * (NT * 16);
  if (n0 >= Np) return;                           // wave-uniform

  const int rowA = lane & 15;
  const int half = lane >> 4;                     // 0: lanes 0-15, 1: lanes 16-31
  // clamp keeps loads in-bounds for a ragged M tail while EXEC stays uniform
  const int rload = (row0 + rowA < M) ? (row0 + rowA) : (M - 1);

  const __bf16* ap = A + (size_t)rload * KP + half * 8;       // A: koff = 0 / 8
  const __bf16* bp[NT];
  #pragma unroll
  for (int t = 0; t < NT; ++t)                                // B: koff = 0 / 8
    bp[t] = Bt + (size_t)(n0 + t * 16 + (lane & 15)) * KP + half * 8;

  v8f acc[NT] = {};
  #pragma unroll
  for (int k0 = 0; k0 < KP; k0 += 32) {
    v8bf alo = *(const v8bf*)(ap + k0);           // K = k0 + koff + 0..7
    v8bf ahi = *(const v8bf*)(ap + k0 + 16);      // K = k0 + 16 + koff + 0..7
    v16bf a = __builtin_shufflevector(alo, ahi, 0, 1, 2, 3, 4, 5, 6, 7,
                                                8, 9, 10, 11, 12, 13, 14, 15);
    if (k0 + 32 < KP) __builtin_prefetch(ap + k0 + 32, 0, 0);
    #pragma unroll
    for (int t = 0; t < NT; ++t) {
      v8bf blo = *(const v8bf*)(bp[t] + k0);      // K = k0 + koff + 0..7
      v8bf bhi = *(const v8bf*)(bp[t] + k0 + 16); // K = k0 + 16 + koff + 0..7
      v16bf b = __builtin_shufflevector(blo, bhi, 0, 1, 2, 3, 4, 5, 6, 7,
                                                  8, 9, 10, 11, 12, 13, 14, 15);
      acc[t] = __builtin_amdgcn_wmma_f32_16x16x32_bf16(
          false, a, false, b, (short)0, acc[t], false, false);
    }
  }

  // C/D layout: VGPR g -> row rbase+g (rbase = 8*laneHalf), lane&15 -> col
  const int colC  = lane & 15;
  const int rbase = half * 8;
  #pragma unroll
  for (int t = 0; t < NT; ++t) {
    #pragma unroll
    for (int g = 0; g < 8; ++g) {
      int r = row0 + rbase + g;
      if (r < M) H[(size_t)r * Np + (n0 + t * 16 + colC)] = acc[t][g];
    }
  }
}

// -------------------------------------------- edge scatter: wave per edge
__global__ void edge_agg(const float* __restrict__ h, const int* __restrict__ src,
                         const int* __restrict__ dst, const float* __restrict__ dinv,
                         float* __restrict__ out, int E, int F, int stride) {
  size_t tid = (size_t)blockIdx.x * blockDim.x + threadIdx.x;
  int wid  = (int)(tid >> 5);
  int lane = threadIdx.x & 31;
  if (wid >= E) return;
  int s = src[wid], d = dst[wid];
  float nrm = dinv[s] * dinv[d];
  const float* hs = h + (size_t)s * stride;
  float* od = out + (size_t)d * stride;
  for (int f = lane; f < F; f += WAVE) atomicAdd(&od[f], nrm * hs[f]);
}

// self-loop term + bias + ReLU (in place).  blockIdx.y = node.
__global__ void finalize_relu(const float* __restrict__ h, const float* __restrict__ dinv,
                              const float* __restrict__ bias, float* __restrict__ out,
                              int F, int stride) {
  int n = blockIdx.y;
  int f = blockIdx.x * blockDim.x + threadIdx.x;
  if (f >= F) return;
  size_t idx = (size_t)n * stride + f;
  float di = dinv[n];
  float v = out[idx] + di * di * h[idx] + bias[f];
  out[idx] = v > 0.0f ? v : 0.0f;
}

// segment max over sorted batch ids; inputs are post-ReLU (>= 0) so the
// uint bit pattern preserves float order and init-to-zero is exact.
// blockIdx.y = node.
__global__ void pool_max(const float* __restrict__ h, const int* __restrict__ batch,
                         unsigned* __restrict__ g, int F, int stride) {
  int n = blockIdx.y;
  int f = blockIdx.x * blockDim.x + threadIdx.x;
  if (f >= F) return;
  atomicMax(&g[(size_t)batch[n] * F + f],
            __float_as_uint(h[(size_t)n * stride + f]));
}

// tiny dense layers for the graph-level MLP.  blockIdx.y = row m.
__global__ void dense(const float* __restrict__ in, const float* __restrict__ Wt,
                      const float* __restrict__ bias, float* __restrict__ out,
                      int K, int Nout, int dorelu) {
  int m = blockIdx.y;
  int n = blockIdx.x * blockDim.x + threadIdx.x;
  if (n >= Nout) return;
  float acc = bias[n];
  for (int k = 0; k < K; ++k) acc += in[(size_t)m * K + k] * Wt[(size_t)k * Nout + n];
  out[(size_t)m * Nout + n] = (dorelu && acc < 0.0f) ? 0.0f : acc;
}

// ---------------------------------------------------------------- launcher
extern "C" void kernel_launch(void* const* d_in, const int* in_sizes, int n_in,
                              void* d_out, int out_size, void* d_ws, size_t ws_size,
                              hipStream_t stream) {
  (void)n_in; (void)ws_size;
  const float* x     = (const float*)d_in[0];
  const int*   ei    = (const int*)d_in[1];
  const int*   batch = (const int*)d_in[2];
  const float* W1 = (const float*)d_in[3];
  const float* b1 = (const float*)d_in[4];
  const float* W2 = (const float*)d_in[5];
  const float* b2 = (const float*)d_in[6];
  const float* Wg = (const float*)d_in[7];
  const float* bg = (const float*)d_in[8];
  const float* Wf = (const float*)d_in[9];
  const float* bf = (const float*)d_in[10];
  const float* Wo = (const float*)d_in[11];
  const float* bo = (const float*)d_in[12];

  const int F1 = 112, F2 = 336, F3 = 168, FG = 84, FF = 42;
  const int K1P = 128, N1P = 384;      // layer-1 padded dims (K%32, N%64)
  const int K2P = 352, N2P = 192;      // layer-2 padded dims
  const int N = in_sizes[0] / F1;
  const int E = in_sizes[1] / 2;
  const int G = out_size;
  const int* src = ei;
  const int* dst = ei + E;

  char* base = (char*)d_ws;
  size_t off = 0;
  auto carveB = [&](size_t bytes) {
    void* p = (void*)(base + off);
    off += ((bytes + 255) & ~(size_t)255);
    return p;
  };
  float*  dinv = (float*)carveB((size_t)N * 4);
  __bf16* Xb   = (__bf16*)carveB((size_t)N * K1P * 2);
  __bf16* W1t  = (__bf16*)carveB((size_t)N1P * K1P * 2);
  float*  h1   = (float*)carveB((size_t)N * N1P * 4);
  float*  a1   = (float*)carveB((size_t)N * N1P * 4);
  __bf16* A1b  = (__bf16*)carveB((size_t)N * K2P * 2);
  __bf16* W2t  = (__bf16*)carveB((size_t)N2P * K2P * 2);
  float*  h2   = (float*)carveB((size_t)N * N2P * 4);
  float*  gbuf = (float*)carveB((size_t)G * F3 * 4);
  float*  g1   = (float*)carveB((size_t)G * FG * 4);
  float*  g2   = (float*)carveB((size_t)G * FF * 4);
  float*  a2   = h1;   // h1 is dead after finalize1 + conversion -> reuse

  const int thr = 256;
  auto cdiv = [](int a, int b) { return (a + b - 1) / b; };

  // degree -> dinv
  zero_f32<<<512, thr, 0, stream>>>(dinv, (size_t)N);
  count_deg<<<cdiv(E, thr), thr, 0, stream>>>(dst, dinv, E);
  make_dinv<<<cdiv(N, thr), thr, 0, stream>>>(dinv, N);

  // ---------------- layer 1: h1 = x @ W1 ; a1 = relu(norm-agg(h1) + b1)
  cvt_pad_bf16<<<dim3(cdiv(K1P, thr), N), thr, 0, stream>>>(x, Xb, F1, F1, K1P);
  transpose_w_bf16<<<dim3(cdiv(K1P, thr), N1P), thr, 0, stream>>>(W1, W1t, F1, F2, K1P);
  {
    int chunks = N1P / (NT * 16);
    dim3 grid(cdiv(N, 16), cdiv(chunks, GEMM_WAVES));
    wmma_gemm_tn<K1P><<<grid, GEMM_WAVES * WAVE, 0, stream>>>(Xb, W1t, h1, N, N1P);
  }
  zero_f32<<<2048, thr, 0, stream>>>(a1, (size_t)N * N1P);
  edge_agg<<<cdiv(E, 8), thr, 0, stream>>>(h1, src, dst, dinv, a1, E, F2, N1P);
  finalize_relu<<<dim3(cdiv(F2, thr), N), thr, 0, stream>>>(h1, dinv, b1, a1, F2, N1P);

  // ---------------- layer 2
  cvt_pad_bf16<<<dim3(cdiv(K2P, thr), N), thr, 0, stream>>>(a1, A1b, F2, N1P, K2P);
  transpose_w_bf16<<<dim3(cdiv(K2P, thr), N2P), thr, 0, stream>>>(W2, W2t, F2, F3, K2P);
  {
    int chunks = N2P / (NT * 16);
    dim3 grid(cdiv(N, 16), cdiv(chunks, GEMM_WAVES));
    wmma_gemm_tn<K2P><<<grid, GEMM_WAVES * WAVE, 0, stream>>>(A1b, W2t, h2, N, N2P);
  }
  zero_f32<<<2048, thr, 0, stream>>>(a2, (size_t)N * N2P);
  edge_agg<<<cdiv(E, 8), thr, 0, stream>>>(h2, src, dst, dinv, a2, E, F3, N2P);
  finalize_relu<<<dim3(cdiv(F3, thr), N), thr, 0, stream>>>(h2, dinv, b2, a2, F3, N2P);

  // ---------------- global max pool + MLP head
  zero_f32<<<256, thr, 0, stream>>>(gbuf, (size_t)G * F3);
  pool_max<<<dim3(cdiv(F3, thr), N), thr, 0, stream>>>(a2, batch, (unsigned*)gbuf, F3, N2P);

  dense<<<dim3(cdiv(FG, 64), G), 64, 0, stream>>>(gbuf, Wg, bg, g1, F3, FG, 1);
  dense<<<dim3(cdiv(FF, 64), G), 64, 0, stream>>>(g1, Wf, bf, g2, FG, FF, 1);
  dense<<<dim3(1, G), 64, 0, stream>>>(g2, Wo, bo, (float*)d_out, FF, 1, 0);
}
